// TriX6502v2_5162550690201
// MI455X (gfx1250) — compile-verified, matrix operand order checked by reading, not run on hardware
//
#include <hip/hip_runtime.h>
#include <hip/hip_bf16.h>

typedef __attribute__((ext_vector_type(16))) __bf16 v16bf;
typedef __attribute__((ext_vector_type(8)))  float  v8f;
typedef unsigned int u32;

#define BTOK 262144

union Frag {
  v16bf v;
  u32 u[8];
};

__device__ __forceinline__ v8f vzero8() {
  v8f z = {0.f,0.f,0.f,0.f,0.f,0.f,0.f,0.f};
  return z;
}

// branch-free gelu (tanh approximation): 0.5x(1+tanh(c(x+0.044715x^3))) == x*sigmoid(2c(x+...))
__device__ __forceinline__ float gelu_fast(float x) {
  float z = -1.5957691216f * (x + 0.044715f * x * x * x);
  return x * __builtin_amdgcn_rcpf(1.f + __expf(z));
}
__device__ __forceinline__ float sigmoid_fast(float x) {
  return __builtin_amdgcn_rcpf(1.f + __expf(-x));
}

// Load a 16x32 (rows x K) bf16 WMMA operand fragment from row-major [rows][K] storage.
// CDNA5 16-bit operand layout: lane = row (&15), halves split K;
// VGPR p<4 -> K = 8*half + 2p ; p>=4 -> K = 16 + 8*half + 2(p-4).
__device__ __forceinline__ void load_frag(Frag& f, const u32* __restrict__ base,
                                          int row, int rowStrideU, int kc, int hf) {
  const u32* p = base + (size_t)row * rowStrideU + kc * 16 + hf * 4;
  f.u[0] = p[0]; f.u[1] = p[1]; f.u[2] = p[2]; f.u[3] = p[3];
  f.u[4] = p[8]; f.u[5] = p[9]; f.u[6] = p[10]; f.u[7] = p[11];
}

// ---------------------------------------------------------------- init
__global__ void k_init(int* counts, int* cursor, float* meanp, float* tern, int* totalp) {
  int t = threadIdx.x;
  if (t < 16) { counts[t] = 0; cursor[t] = 0; meanp[t] = 0.f; }
  if (t == 0) { *tern = 0.f; *totalp = 0; }
}

// ------------------------------------------------- ternary quant (transposed bf16 out)
__global__ __launch_bounds__(256)
void k_quant(const float* __restrict__ W1, const float* __restrict__ W2,
             __bf16* __restrict__ W1qT, __bf16* __restrict__ W2qT, float* tern) {
  int bid = blockIdx.x, tid = threadIdx.x;
  const float* W; __bf16* dst; int rows, cshift, cmask;
  if (bid < 16) { W = W1 + (size_t)bid * 65536; dst = W1qT + (size_t)bid * 65536; rows = 128; cshift = 9; cmask = 511; }
  else          { W = W2 + (size_t)(bid - 16) * 65536; dst = W2qT + (size_t)(bid - 16) * 65536; rows = 512; cshift = 7; cmask = 127; }
  __shared__ float red[256];
  float s = 0.f;
  for (int i = tid; i < 65536; i += 256) s += fabsf(W[i]);
  red[tid] = s; __syncthreads();
  for (int o = 128; o > 0; o >>= 1) { if (tid < o) red[tid] += red[tid + o]; __syncthreads(); }
  float s0 = red[0] * (1.f / 65536.f);
  __syncthreads();
  float thr = 0.7f * s0, ls = 0.f;
  for (int i = tid; i < 65536; i += 256) {
    float w = W[i];
    float q = (fabsf(w) > thr) ? ((w > 0.f) ? s0 : -s0) : 0.f;
    int r = i >> cshift, c = i & cmask;
    dst[(size_t)c * rows + r] = (__bf16)q;   // transposed: [col][row]
    ls += fabsf(w - q);
  }
  red[tid] = ls; __syncthreads();
  for (int o = 128; o > 0; o >>= 1) { if (tid < o) red[tid] += red[tid + o]; __syncthreads(); }
  if (tid == 0) atomicAdd(tern, red[0] * (1.f / 1048576.f));
}

// ------------------------------------------------- head weights -> transposed bf16
__global__ void k_head_conv(const float* Wh1, const float* Wh2, __bf16* Wh1T, __bf16* Wh2T) {
  int i = blockIdx.x * 256 + threadIdx.x;
  if (i < 128 * 64) { int r = i >> 6, c = i & 63; Wh1T[c * 128 + r] = (__bf16)Wh1[i]; }
  if (i < 64 * 8)   { int r = i >> 3, c = i & 7;  Wh2T[c * 64 + r]  = (__bf16)Wh2[i]; }
}

// ------------------------------------------------- front-end: features, x, router
__global__ __launch_bounds__(256)
void k_front(const int* __restrict__ op_idx, const int* __restrict__ aa,
             const int* __restrict__ bb, const int* __restrict__ cc,
             const float* __restrict__ op_embed, const float* __restrict__ W_in,
             const float* __restrict__ b_in, const float* __restrict__ Wr,
             __bf16* __restrict__ xbuf, float* __restrict__ gatebuf,
             int* __restrict__ idxbuf, int* __restrict__ counts,
             float* __restrict__ meanp, float* __restrict__ out) {
  __shared__ float sWin[49 * 128];
  __shared__ float sWr[128 * 16];
  __shared__ float sEmb[8 * 32];
  __shared__ float sBin[128];
  __shared__ float sRed[16];
  __shared__ int   sCnt[16];
  int tid = threadIdx.x;
  for (int i = tid; i < 49 * 128; i += 256) sWin[i] = W_in[i];
  for (int i = tid; i < 128 * 16; i += 256) sWr[i] = Wr[i];
  for (int i = tid; i < 8 * 32; i += 256)   sEmb[i] = op_embed[i];
  if (tid < 128) sBin[tid] = b_in[tid];
  if (tid < 16) { sRed[tid] = 0.f; sCnt[tid] = 0; }
  __syncthreads();

  int tok = blockIdx.x * 256 + tid;
  int op = op_idx[tok], av = aa[tok], bv = bb[tok];
  float cv = (float)cc[tok];
  float embr[32];
#pragma unroll
  for (int e = 0; e < 32; ++e) embr[e] = sEmb[op * 32 + e];
  float logits[16];
#pragma unroll
  for (int j = 0; j < 16; ++j) logits[j] = 0.f;
  u32* xrow = (u32*)(xbuf + (size_t)tok * 128);
  for (int d = 0; d < 128; d += 2) {
    float a0 = sBin[d], a1 = sBin[d + 1];
    const float* w = sWin + d;
#pragma unroll
    for (int e = 0; e < 32; ++e) { float ev = embr[e]; a0 += ev * w[e * 128]; a1 += ev * w[e * 128 + 1]; }
#pragma unroll
    for (int i = 0; i < 8; ++i) if ((av >> i) & 1) { a0 += w[(32 + i) * 128]; a1 += w[(32 + i) * 128 + 1]; }
#pragma unroll
    for (int i = 0; i < 8; ++i) if ((bv >> i) & 1) { a0 += w[(40 + i) * 128]; a1 += w[(40 + i) * 128 + 1]; }
    a0 += cv * w[48 * 128]; a1 += cv * w[48 * 128 + 1];
    union { __bf16 h[2]; u32 u; } pk;
    pk.h[0] = (__bf16)a0; pk.h[1] = (__bf16)a1;
    xrow[d >> 1] = pk.u;
    const float* wr = sWr + d * 16;
#pragma unroll
    for (int j = 0; j < 16; ++j) logits[j] += a0 * wr[j] + a1 * wr[16 + j];
  }
  float mx = logits[0]; int best = 0;
#pragma unroll
  for (int j = 1; j < 16; ++j) if (logits[j] > mx) { mx = logits[j]; best = j; }
  float pr[16], sum = 0.f;
#pragma unroll
  for (int j = 0; j < 16; ++j) { pr[j] = __expf(logits[j] - mx); sum += pr[j]; }
  float inv = 1.f / sum;
  float gate = inv;  // pr[best] == 1 -> max prob = 1/sum
  gatebuf[tok] = gate;
  idxbuf[tok] = best;
  out[(size_t)BTOK * 8 + tok] = (float)best;  // idx output
#pragma unroll
  for (int j = 0; j < 16; ++j) atomicAdd(&sRed[j], pr[j] * inv);
  atomicAdd(&sCnt[best], 1);
  __syncthreads();
  if (tid < 16) { atomicAdd(&meanp[tid], sRed[tid]); atomicAdd(&counts[tid], sCnt[tid]); }
}

// ------------------------------------------------- offsets, group descriptors (32 tok), aux
__global__ __launch_bounds__(256)
void k_finalize(const int* counts, const float* meanp, const float* tern,
                int* offsets, int* totalp, int* gdesc, float* out) {
  __shared__ int sGbase[17];
  __shared__ int sTotal;
  int tid = threadIdx.x;
  if (tid == 0) {
    int off = 0, gb = 0;
    for (int t = 0; t < 16; ++t) {
      offsets[t] = off; off += counts[t];
      sGbase[t] = gb; gb += (counts[t] + 31) >> 5;
    }
    offsets[16] = off; sGbase[16] = gb; sTotal = gb; *totalp = gb;
    float sp = 0.f, cp[4] = {0.f, 0.f, 0.f, 0.f};
    const float invB = 1.f / (float)BTOK;
    for (int t = 0; t < 16; ++t) {
      float mp = meanp[t] * invB;
      float fr = (float)counts[t] * invB;
      sp += fr * mp;
      cp[t >> 2] += mp;
    }
    sp *= 16.f;
    float dv = 0.f;
    for (int c4 = 0; c4 < 4; ++c4) dv += cp[c4] * logf(cp[c4] + 1e-9f);
    out[(size_t)BTOK * 9] = 0.01f * (*tern) + 0.005f * sp + 0.01f * dv;  // aux
  }
  __syncthreads();
  int total = sTotal;
  for (int e = tid; e < total; e += 256) {
    int t = 0;
    while (e >= sGbase[t + 1]) t++;
    gdesc[e] = (t << 20) | (e - sGbase[t]);
  }
}

// ------------------------------------------------- scatter tokens per tile
__global__ __launch_bounds__(256)
void k_scatter(const int* idxbuf, const int* offsets, int* cursor, int* tlist) {
  int tok = blockIdx.x * 256 + threadIdx.x;
  int t = idxbuf[tok];
  int pos = atomicAdd(&cursor[t], 1);
  tlist[offsets[t] + pos] = tok;
}

// ------------------------------------------------- WMMA FFN + head, one wave / 32 tokens
#define XT_OFF   0        // 32 x 128 bf16 = 8 KB
#define HT_OFF   8192     // 32 x 512 bf16 = 32 KB
#define TOK_OFF  40960
#define GATE_OFF 41088
#define SMEM_SZ  41216

__global__ __launch_bounds__(32)
void k_ffn(const __bf16* __restrict__ xbuf, const float* __restrict__ gatebuf,
           const __bf16* __restrict__ W1qT, const __bf16* __restrict__ W2qT,
           const __bf16* __restrict__ Wh1T, const __bf16* __restrict__ Wh2T,
           const float* __restrict__ bh1, const float* __restrict__ bh2,
           const int* __restrict__ counts, const int* __restrict__ offsets,
           const int* __restrict__ tlist, const int* __restrict__ gdesc,
           const int* __restrict__ totalp, float* __restrict__ out) {
  if ((int)blockIdx.x >= *totalp) return;
  int desc = gdesc[blockIdx.x];
  int t = desc >> 20, gi = desc & 0xFFFFF;
  int lane = threadIdx.x, m = lane & 15, hf = lane >> 4;

  __shared__ __align__(16) unsigned char smem[SMEM_SZ];
  u32*    xT    = (u32*)(smem + XT_OFF);
  __bf16* hT    = (__bf16*)(smem + HT_OFF);
  int*    toks  = (int*)(smem + TOK_OFF);
  float*  gateS = (float*)(smem + GATE_OFF);

  int cnt = counts[t];
  int validCnt = cnt - (gi << 5); if (validCnt > 32) validCnt = 32;
  int base = offsets[t] + (gi << 5);
  {
    int src = (lane < validCnt) ? lane : 0;
    int tk = tlist[base + src];
    toks[lane] = tk;
    gateS[lane] = gatebuf[tk];
  }
  __syncthreads();
  {  // stage 32 x-rows (each lane copies one full 256B row)
    const u32* xr = (const u32*)(xbuf + (size_t)toks[lane] * 128);
    u32* xd = xT + lane * 64;
#pragma unroll
    for (int i = 0; i < 64; i += 4) { xd[i] = xr[i]; xd[i+1] = xr[i+1]; xd[i+2] = xr[i+2]; xd[i+3] = xr[i+3]; }
  }
  __syncthreads();

  const u32* W1u = (const u32*)(W1qT + (size_t)t * 65536);  // [512][128] rows = 64 uints
  const u32* W2u = (const u32*)(W2qT + (size_t)t * 65536);  // [128][512] rows = 256 uints
  int n = m;

  // GEMM1: h = gelu(x @ W1)   (32x128)*(128x512): 2 M-tiles share each B fragment
  Frag A[2][4];
#pragma unroll
  for (int mt = 0; mt < 2; ++mt)
#pragma unroll
    for (int kc = 0; kc < 4; ++kc) load_frag(A[mt][kc], xT, mt * 16 + m, 64, kc, hf);
  for (int nt = 0; nt < 32; ++nt) {
    v8f acc[2]; acc[0] = vzero8(); acc[1] = vzero8();
#pragma unroll
    for (int kc = 0; kc < 4; ++kc) {
      Frag Bf; load_frag(Bf, W1u, nt * 16 + n, 64, kc, hf);
#pragma unroll
      for (int mt = 0; mt < 2; ++mt)
        acc[mt] = __builtin_amdgcn_wmma_f32_16x16x32_bf16(false, A[mt][kc].v, false, Bf.v, (short)0, acc[mt], false, false);
    }
#pragma unroll
    for (int mt = 0; mt < 2; ++mt)
#pragma unroll
      for (int r = 0; r < 8; ++r)
        hT[(mt * 16 + r + 8 * hf) * 512 + nt * 16 + n] = (__bf16)gelu_fast(acc[mt][r]);
  }
  __syncthreads();

  // GEMM2: y = h @ W2   (32x512)*(512x128): hold 2x8 accumulators, share B fragments
  v8f acc2[2][8];
#pragma unroll
  for (int mt = 0; mt < 2; ++mt)
#pragma unroll
    for (int i = 0; i < 8; ++i) acc2[mt][i] = vzero8();
  const u32* hU = (const u32*)hT;
  for (int kc = 0; kc < 16; ++kc) {
    Frag A2[2];
#pragma unroll
    for (int mt = 0; mt < 2; ++mt) load_frag(A2[mt], hU, mt * 16 + m, 256, kc, hf);
#pragma unroll
    for (int nt = 0; nt < 8; ++nt) {
      Frag Bf; load_frag(Bf, W2u, nt * 16 + n, 256, kc, hf);
#pragma unroll
      for (int mt = 0; mt < 2; ++mt)
        acc2[mt][nt] = __builtin_amdgcn_wmma_f32_16x16x32_bf16(false, A2[mt].v, false, Bf.v, (short)0, acc2[mt][nt], false, false);
    }
  }
  __syncthreads();
  __bf16* oT = (__bf16*)(smem + HT_OFF);  // reuse hT region: 32 x 128 bf16
#pragma unroll
  for (int mt = 0; mt < 2; ++mt)
#pragma unroll
    for (int nt = 0; nt < 8; ++nt)
#pragma unroll
      for (int r = 0; r < 8; ++r) {
        float v = acc2[mt][nt][r] * gateS[mt * 16 + r + 8 * hf];
        oT[(mt * 16 + r + 8 * hf) * 128 + nt * 16 + n] = (__bf16)v;
      }
  __syncthreads();

  // Head 1: h1 = relu(y @ Wh1 + bh1)   (32x128)*(128x64)
  const u32* oU = (const u32*)oT;
  const u32* Wh1u = (const u32*)Wh1T;   // [64][128] rows = 64 uints
  Frag A3[2][4];
#pragma unroll
  for (int mt = 0; mt < 2; ++mt)
#pragma unroll
    for (int kc = 0; kc < 4; ++kc) load_frag(A3[mt][kc], oU, mt * 16 + m, 64, kc, hf);
  __bf16* h1T = (__bf16*)(smem + XT_OFF);  // reuse xT region: 32 x 64 bf16
#pragma unroll
  for (int nt = 0; nt < 4; ++nt) {
    v8f acc[2]; acc[0] = vzero8(); acc[1] = vzero8();
#pragma unroll
    for (int kc = 0; kc < 4; ++kc) {
      Frag Bf; load_frag(Bf, Wh1u, nt * 16 + n, 64, kc, hf);
#pragma unroll
      for (int mt = 0; mt < 2; ++mt)
        acc[mt] = __builtin_amdgcn_wmma_f32_16x16x32_bf16(false, A3[mt][kc].v, false, Bf.v, (short)0, acc[mt], false, false);
    }
    float bias = bh1[nt * 16 + n];
#pragma unroll
    for (int mt = 0; mt < 2; ++mt)
#pragma unroll
      for (int r = 0; r < 8; ++r) {
        float v = fmaxf(acc[mt][r] + bias, 0.f);
        h1T[(mt * 16 + r + 8 * hf) * 64 + nt * 16 + n] = (__bf16)v;
      }
  }
  __syncthreads();

  // Head 2: res = sigmoid(h1 @ Wh2 + bh2)   (32x64)*(64x8)
  const u32* h1U = (const u32*)h1T;
  const u32* Wh2u = (const u32*)Wh2T;   // [8][64] rows = 32 uints
  int n2 = (n < 8) ? n : 0;
  v8f accR[2]; accR[0] = vzero8(); accR[1] = vzero8();
#pragma unroll
  for (int kc = 0; kc < 2; ++kc) {
    Frag Bf; load_frag(Bf, Wh2u, n2, 32, kc, hf);
#pragma unroll
    for (int mt = 0; mt < 2; ++mt) {
      Frag A4; load_frag(A4, h1U, mt * 16 + m, 32, kc, hf);
      accR[mt] = __builtin_amdgcn_wmma_f32_16x16x32_bf16(false, A4.v, false, Bf.v, (short)0, accR[mt], false, false);
    }
  }
  if (n < 8) {
    float bias = bh2[n];
#pragma unroll
    for (int mt = 0; mt < 2; ++mt)
#pragma unroll
      for (int r = 0; r < 8; ++r) {
        int mr = mt * 16 + r + 8 * hf;
        if (mr < validCnt) {
          float v = sigmoid_fast(accR[mt][r] + bias);
          out[(size_t)toks[mr] * 8 + n] = v;
        }
      }
  }
}

// ---------------------------------------------------------------- launch
extern "C" void kernel_launch(void* const* d_in, const int* in_sizes, int n_in,
                              void* d_out, int out_size, void* d_ws, size_t ws_size,
                              hipStream_t stream) {
  const int*   op_idx  = (const int*)d_in[0];
  const int*   a       = (const int*)d_in[1];
  const int*   b       = (const int*)d_in[2];
  const int*   c       = (const int*)d_in[3];
  const float* op_embed= (const float*)d_in[4];
  const float* W_in    = (const float*)d_in[5];
  const float* b_in    = (const float*)d_in[6];
  const float* Wr      = (const float*)d_in[7];
  const float* W1      = (const float*)d_in[8];
  const float* W2      = (const float*)d_in[9];
  const float* Wh1     = (const float*)d_in[10];
  const float* bh1     = (const float*)d_in[11];
  const float* Wh2     = (const float*)d_in[12];
  const float* bh2     = (const float*)d_in[13];
  float* out = (float*)d_out;

  unsigned char* ws = (unsigned char*)d_ws;
  size_t o = 0;
  auto take = [&](size_t bytes) {
    unsigned char* p = ws + o;
    o = (o + bytes + 255) & ~(size_t)255;
    return p;
  };
  __bf16* W1qT   = (__bf16*)take(2097152);
  __bf16* W2qT   = (__bf16*)take(2097152);
  __bf16* xbuf   = (__bf16*)take((size_t)BTOK * 128 * 2);
  __bf16* Wh1T   = (__bf16*)take(16384);
  __bf16* Wh2T   = (__bf16*)take(1024);
  float*  gatebuf= (float*)take((size_t)BTOK * 4);
  int*    idxbuf = (int*)take((size_t)BTOK * 4);
  int*    tlist  = (int*)take((size_t)BTOK * 4);
  int*    counts = (int*)take(64);
  int*    cursor = (int*)take(64);
  int*    offsets= (int*)take(128);
  float*  meanp  = (float*)take(64);
  float*  tern   = (float*)take(256);
  int*    totalp = (int*)take(256);
  int*    gdesc  = (int*)take(8320 * 4);

  k_init<<<1, 64, 0, stream>>>(counts, cursor, meanp, tern, totalp);
  k_quant<<<32, 256, 0, stream>>>(W1, W2, W1qT, W2qT, tern);
  k_head_conv<<<32, 256, 0, stream>>>(Wh1, Wh2, Wh1T, Wh2T);
  k_front<<<BTOK / 256, 256, 0, stream>>>(op_idx, a, b, c, op_embed, W_in, b_in, Wr,
                                          xbuf, gatebuf, idxbuf, counts, meanp, out);
  k_finalize<<<1, 256, 0, stream>>>(counts, meanp, tern, offsets, totalp, gdesc, out);
  k_scatter<<<BTOK / 256, 256, 0, stream>>>(idxbuf, offsets, cursor, tlist);
  k_ffn<<<8320, 32, 0, stream>>>(xbuf, gatebuf, W1qT, W2qT, Wh1T, Wh2T, bh1, bh2,
                                 counts, offsets, tlist, gdesc, totalp, out);
}
